// Attention_6502580486519
// MI455X (gfx1250) — matrix operand order repaired and checked by reading, not compiled, not run on hardware
//
#include <hip/hip_runtime.h>

typedef __attribute__((ext_vector_type(16))) _Float16 v16h;
typedef __attribute__((ext_vector_type(8)))  _Float16 v8h;
typedef __attribute__((ext_vector_type(8)))  float    v8f;

#define B_ 4
#define S_ 4096
#define E_ 1024
#define L_ 64
#define M_ (B_*S_)   // 16384 total rows

// ---------------------------------------------------------------------------
// Kernel 0: transpose W [E,L] f32 -> Wt [L,E] f16 (x3 matrices) so the WMMA
// B-operand (fixed column n, 16 consecutive k per lane) is a contiguous load.
// ---------------------------------------------------------------------------
__global__ void wtrans(const float* __restrict__ Wq, const float* __restrict__ Wk,
                       const float* __restrict__ Wv, _Float16* __restrict__ Wt)
{
    int t   = blockIdx.x * blockDim.x + threadIdx.x;   // 0 .. 3*E*L-1
    int mat = t / (E_ * L_);
    int r   = t - mat * (E_ * L_);
    int n   = r / E_;
    int e   = r - n * E_;
    const float* Wsrc = (mat == 0) ? Wq : ((mat == 1) ? Wk : Wv);
    Wt[(size_t)mat * (E_ * L_) + (size_t)n * E_ + e] = (_Float16)Wsrc[(size_t)e * L_ + n];
}

// ---------------------------------------------------------------------------
// Kernel 1: QKV projection, all WMMA f32_16x16x32_f16.
// grid = (M/16, 3), block = 128 (4 waves, one 16-wide n-tile each).
// Q is pre-scaled by 1/sqrt(L); V is stored transposed [B, L, S] f16.
// ---------------------------------------------------------------------------
__global__ void __launch_bounds__(128)
qkv_proj(const float* __restrict__ x, const _Float16* __restrict__ Wt3,
         const float* __restrict__ bq, const float* __restrict__ bk,
         const float* __restrict__ bv,
         _Float16* __restrict__ Qh, _Float16* __restrict__ Kh,
         _Float16* __restrict__ Vt)
{
    const int lane = threadIdx.x & 31;
    const int wave = threadIdx.x >> 5;
    const int half = lane >> 4;
    const int l15  = lane & 15;
    const int tile = blockIdx.x;          // 16-row tile
    const int mat  = blockIdx.y;          // 0=Q 1=K 2=V
    const int n0   = wave * 16;

    const _Float16* Wt   = Wt3 + (size_t)mat * (E_ * (size_t)L_);
    const float*    bias = (mat == 0) ? bq : ((mat == 1) ? bk : bv);

    const int    row  = tile * 16 + l15;                  // A-row for this lane
    const float* xrow = x + (size_t)row * E_;

    v8f acc = {};
    for (int kk = 0; kk < E_; kk += 32) {
        // A 16x32: lane holds row (lane&15), k = 16h + 0..15
        v8f a0 = *(const v8f*)(xrow + kk + 16 * half);
        v8f a1 = *(const v8f*)(xrow + kk + 16 * half + 8);
        v16h a;
#pragma unroll
        for (int i = 0; i < 8; ++i) { a[i] = (_Float16)a0[i]; a[8 + i] = (_Float16)a1[i]; }
        // B 32x16: lane holds col (lane&15), k = 16h + 0..15 (contiguous in Wt)
        v16h bm = *(const v16h*)(Wt + (size_t)(n0 + l15) * E_ + kk + 16 * half);
        acc = __builtin_amdgcn_wmma_f32_16x16x32_f16(false, a, false, bm,
                                                     (short)0, acc, false, false);
    }

    const int   ng = n0 + l15;
    const float bb = bias[ng];
    if (mat == 2) {
        // Vt[b, n, s]: lane stores 8 consecutive s as one 16B store
        const int r0 = tile * 16 + 8 * half;
        const int b  = r0 / S_;
        const int s0 = r0 - b * S_;
        v8h v;
#pragma unroll
        for (int i = 0; i < 8; ++i) v[i] = (_Float16)(acc[i] + bb);
        *(v8h*)(Vt + ((size_t)b * L_ + ng) * S_ + s0) = v;
    } else {
        _Float16*   dst = (mat == 0) ? Qh : Kh;
        const float sc  = (mat == 0) ? 0.125f : 1.0f;     // fold 1/sqrt(64) into Q
#pragma unroll
        for (int i = 0; i < 8; ++i) {
            int r = tile * 16 + 8 * half + i;
            dst[(size_t)r * L_ + ng] = (_Float16)((acc[i] + bb) * sc);
        }
    }
}

// ---------------------------------------------------------------------------
// Kernel 2: flash attention. 1 wave = one 32-query tile (two 16-row A tiles
// sharing each K/V block -> 2x arithmetic intensity vs 16 q/wave).
// Streams key blocks of 32 with online softmax; scores & output accumulators
// live in VGPRs. block = 128 (4 waves), grid = M/128.
// ---------------------------------------------------------------------------
__global__ void __launch_bounds__(128)
attn(const _Float16* __restrict__ Qh, const _Float16* __restrict__ Kh,
     const _Float16* __restrict__ Vt, float* __restrict__ out)
{
    __shared__ _Float16 plds[4][16 * 32];     // per-wave P staging (D->A transpose)
    const int lane = threadIdx.x & 31;
    const int wave = threadIdx.x >> 5;
    const int half = lane >> 4;
    const int l15  = lane & 15;
    const int qt   = blockIdx.x * 4 + wave;   // global 32-query tile id (0..511)
    const int q0   = qt * 32;                 // first global row of this wave
    const int b    = q0 / S_;
    const int s0   = q0 - b * S_;             // tile start within batch
    _Float16* P    = plds[wave];

    // Q A-tiles: 2 row-tiles x 2 k-chunks (16x32 each), loaded once
    v16h Qa[2][2];
#pragma unroll
    for (int t = 0; t < 2; ++t) {
        const _Float16* qbase = Qh + ((size_t)q0 + t * 16 + l15) * L_;
#pragma unroll
        for (int c = 0; c < 2; ++c) {
            v8h lo = *(const v8h*)(qbase + c * 32 + 16 * half);
            v8h hi = *(const v8h*)(qbase + c * 32 + 16 * half + 8);
#pragma unroll
            for (int i = 0; i < 8; ++i) { Qa[t][c][i] = lo[i]; Qa[t][c][8 + i] = hi[i]; }
        }
    }

    float rm[2][8], rs[2][8];
    v8f   acc[2][4];
#pragma unroll
    for (int t = 0; t < 2; ++t) {
#pragma unroll
        for (int i = 0; i < 8; ++i) { rm[t][i] = -__builtin_inff(); rs[t][i] = 0.0f; }
#pragma unroll
        for (int n4 = 0; n4 < 4; ++n4) { v8f z = {}; acc[t][n4] = z; }
    }

    const int qlast = s0 + 31;
    for (int j0 = 0; j0 <= qlast; j0 += 32) {
        // ---- scores: 32 queries x 32 keys; K tiles loaded once, used twice ----
        v8f sc[2][2];                          // [row-tile][key n-tile]
#pragma unroll
        for (int nt = 0; nt < 2; ++nt) {
            const _Float16* kbase = Kh + ((size_t)b * S_ + j0 + nt * 16 + l15) * L_;
            v16h kb0 = *(const v16h*)(kbase + 16 * half);
            v16h kb1 = *(const v16h*)(kbase + 32 + 16 * half);
#pragma unroll
            for (int t = 0; t < 2; ++t) {
                v8f z = {};
                v8f s = __builtin_amdgcn_wmma_f32_16x16x32_f16(false, Qa[t][0], false, kb0,
                                                               (short)0, z, false, false);
                s = __builtin_amdgcn_wmma_f32_16x16x32_f16(false, Qa[t][1], false, kb1,
                                                           (short)0, s, false, false);
                sc[t][nt] = s;
            }
        }

#pragma unroll
        for (int t = 0; t < 2; ++t) {
            const int sq = s0 + t * 16;                    // row-tile start in batch
            // ---- causal mask (only on blocks straddling/above this tile) ----
            if (j0 + 31 > sq) {
#pragma unroll
                for (int nt = 0; nt < 2; ++nt) {
                    int key = j0 + nt * 16 + l15;
#pragma unroll
                    for (int i = 0; i < 8; ++i) {
                        int q = sq + 8 * half + i;         // D-layout row
                        sc[t][nt][i] = (key > q) ? -__builtin_inff() : sc[t][nt][i];
                    }
                }
            }
            // ---- online softmax (row reductions across 16-lane groups) ----
            float alpha[8];
#pragma unroll
            for (int i = 0; i < 8; ++i) {
                float mx = fmaxf(sc[t][0][i], sc[t][1][i]);
#pragma unroll
                for (int m = 1; m < 16; m <<= 1) mx = fmaxf(mx, __shfl_xor(mx, m, 32));
                float nm = fmaxf(rm[t][i], mx);
                alpha[i] = __expf(rm[t][i] - nm);
                rm[t][i] = nm;
                float p0 = __expf(sc[t][0][i] - nm);
                float p1 = __expf(sc[t][1][i] - nm);
                sc[t][0][i] = p0; sc[t][1][i] = p1;
                float ps = p0 + p1;
#pragma unroll
                for (int m = 1; m < 16; m <<= 1) ps += __shfl_xor(ps, m, 32);
                rs[t][i] = rs[t][i] * alpha[i] + ps;
            }
#pragma unroll
            for (int n4 = 0; n4 < 4; ++n4)
#pragma unroll
                for (int i = 0; i < 8; ++i) acc[t][n4][i] *= alpha[i];

            // ---- P: D-layout -> A-layout via per-wave LDS slab ----
#pragma unroll
            for (int nt = 0; nt < 2; ++nt)
#pragma unroll
                for (int i = 0; i < 8; ++i)
                    P[(8 * half + i) * 32 + nt * 16 + l15] = (_Float16)sc[t][nt][i];
            asm volatile("s_wait_dscnt 0" ::: "memory");   // CDNA5 split DS counter
            v16h pa;
            {
                v8h lo = *(const v8h*)(P + l15 * 32 + 16 * half);
                v8h hi = *(const v8h*)(P + l15 * 32 + 16 * half + 8);
#pragma unroll
                for (int i = 0; i < 8; ++i) { pa[i] = lo[i]; pa[8 + i] = hi[i]; }
            }
            // ---- O += P (16x32) * V (32x64): 4 n-tiles, V from transposed Vt ----
#pragma unroll
            for (int n4 = 0; n4 < 4; ++n4) {
                v16h vb = *(const v16h*)(Vt + ((size_t)b * L_ + n4 * 16 + l15) * S_
                                            + j0 + 16 * half);
                acc[t][n4] = __builtin_amdgcn_wmma_f32_16x16x32_f16(false, pa, false, vb,
                                                                    (short)0, acc[t][n4],
                                                                    false, false);
            }
        }
    }

    // ---- epilogue: normalize and store fp32 ----
#pragma unroll
    for (int t = 0; t < 2; ++t)
#pragma unroll
        for (int n4 = 0; n4 < 4; ++n4)
#pragma unroll
            for (int i = 0; i < 8; ++i) {
                size_t r = (size_t)q0 + t * 16 + 8 * half + i;
                out[r * L_ + n4 * 16 + l15] = acc[t][n4][i] / rs[t][i];
            }
}

// ---------------------------------------------------------------------------
extern "C" void kernel_launch(void* const* d_in, const int* in_sizes, int n_in,
                              void* d_out, int out_size, void* d_ws, size_t ws_size,
                              hipStream_t stream)
{
    const float* x  = (const float*)d_in[0];
    const float* Wq = (const float*)d_in[1];
    const float* Wk = (const float*)d_in[2];
    const float* Wv = (const float*)d_in[3];
    const float* bq = (const float*)d_in[4];
    const float* bk = (const float*)d_in[5];
    const float* bv = (const float*)d_in[6];
    // d_in[7] = causal mask (implicit in kernel; unused)

    char* ws = (char*)d_ws;
    _Float16* Wt = (_Float16*)ws;                                   // 3*L*E f16 = 384 KB
    _Float16* Qh = (_Float16*)(ws + 0x60000);                       // M*L f16  = 2 MB
    _Float16* Kh = (_Float16*)(ws + 0x60000 + 0x200000);            // 2 MB
    _Float16* Vt = (_Float16*)(ws + 0x60000 + 0x400000);            // 2 MB (transposed)
    float* out = (float*)d_out;

    wtrans  <<<dim3((3 * E_ * L_) / 256), dim3(256), 0, stream>>>(Wq, Wk, Wv, Wt);
    qkv_proj<<<dim3(M_ / 16, 3),          dim3(128), 0, stream>>>(x, Wt, bq, bk, bv,
                                                                  Qh, Kh, Vt);
    attn    <<<dim3(M_ / 128),            dim3(128), 0, stream>>>(Qh, Kh, Vt, out);
}
